// SNNConv2d_42219528519842
// MI455X (gfx1250) — matrix operand order repaired and checked by reading, not compiled, not run on hardware
//
#include <hip/hip_runtime.h>

// ---------------------------------------------------------------------------
// SNN Conv2d (3x3, pad 1) + integrate-fire scan, fused, for MI455X (gfx1250).
// conv as implicit GEMM with v_wmma_f32_16x16x32_bf16; membrane state lives in
// VGPRs across the T-loop; WMMA C is seeded with mem+bias each step.
// Preferred path: spikes pre-transposed to bf16 [t][b][y][x][cin] in d_ws, then
// staged per-timestep via global_load_async_to_lds_b128 with double buffering
// (ASYNCcnt). Fallback path (small d_ws): direct f32 loads + VALU cvt.
// ---------------------------------------------------------------------------

typedef __bf16 bf16;
typedef __attribute__((ext_vector_type(16))) __bf16 v16bf;
typedef __attribute__((ext_vector_type(8)))  float  v8f;

#define T_    50
#define B_    8
#define CIN   64
#define COUT  128
#define H_    32
#define W_    32
#define VTH   15.0f

#define TILE_H 2                 // output rows per workgroup
#define TILE_M 64                // TILE_H * W_
#define TILE_N 64                // couts per workgroup (half of COUT)
#define HALO_H (TILE_H + 2)      // 4
#define HALO_W (W_ + 2)          // 34
#define KSTEPS 18                // 9 taps * 2 cin-halves (K=32 per WMMA)
#define SOUT_STRIDE 68           // 64 + 4 pad floats: kills LDS bank conflicts
#define A_ELEMS (HALO_H * HALO_W * CIN)   // 8704 bf16 per tile buffer

// wsB: [tap(9)][kHalf(2)][nTile(8)][lane(32)][slot(16)] bf16
#define WSB_ELEMS (9 * 2 * 8 * 32 * 16)               // 73728 bf16
// wsS: [t][b][y][x][cin] bf16
#define WSS_ELEMS ((size_t)T_ * B_ * H_ * W_ * CIN)   // 26,214,400 bf16

// ---------------------------------------------------------------------------
// Pack f32 weights into per-lane WMMA B fragments (bf16).
// A slots are loaded contiguously (cin = base + 16*half + j); the implied
// K-permutation pi(hwK) = 16*((hwK>>3)&1) + (hwK&7) + 8*(hwK>>4) is applied
// identically here so the WMMA dot products are consistent.
// ---------------------------------------------------------------------------
__global__ void snn_pack_weights(const float* __restrict__ w,
                                 bf16* __restrict__ wsB) {
  int e = blockIdx.x * 256 + threadIdx.x;
  if (e >= WSB_ELEMS) return;
  int j     = e & 15;
  int lane  = (e >> 4) & 31;
  int nTile = (e >> 9) & 7;
  int kHalf = (e >> 12) & 1;
  int tap   = e >> 13;                       // 0..8
  int kh = tap / 3, kw = tap % 3;
  int halfB = lane >> 4;
  int n     = nTile * 16 + (lane & 15);
  int hwK   = halfB * 16 + j;
  int cin   = kHalf * 32 + hwK;
  float v = w[((n * CIN + cin) * 3 + kh) * 3 + kw];
  wsB[e] = (bf16)v;
}

// ---------------------------------------------------------------------------
// Transpose + convert spikes: [t][b][cin][y][x] f32 -> [t][b][y][x][cin] bf16.
// One block per (t*B+b, y) row. Reads coalesced over x, writes coalesced 16B.
// ---------------------------------------------------------------------------
__global__ __launch_bounds__(256) void snn_transpose_spikes(
    const float* __restrict__ sp, bf16* __restrict__ wsS) {
  __shared__ bf16 tile[32 * 80];            // stride 80 keeps uint4 alignment
  int row = blockIdx.x;                     // T_*B_*H_ rows
  int tb  = row >> 5;
  int y   = row & 31;
  const float* src = sp + (size_t)tb * CIN * H_ * W_ + (size_t)y * W_;
  int x  = threadIdx.x & 31;
  int c0 = threadIdx.x >> 5;                // wave-uniform channel group
  for (int i = 0; i < 8; ++i) {
    int cin = i * 8 + c0;
    tile[x * 80 + cin] = (bf16)src[(size_t)cin * (H_ * W_) + x];
  }
  __syncthreads();
  int xo = threadIdx.x >> 3, cb = threadIdx.x & 7;
  uint4 v = *(const uint4*)&tile[xo * 80 + cb * 8];
  *(uint4*)&wsS[((size_t)tb * H_ + y) * (W_ * CIN) + xo * CIN + cb * 8] = v;
}

// ---------------------------------------------------------------------------
// Async-staged fused conv + scan. Grid: 256 WGs = B(8) x tileY(16) x nHalf(2).
// ---------------------------------------------------------------------------
__global__ __launch_bounds__(256) void snn_conv_scan_async(
    const bf16*  __restrict__ wsS,      // transposed bf16 spikes
    const float* __restrict__ mem_in,   // [B][COUT][H][W]
    const float* __restrict__ bias,     // [COUT]
    const bf16*  __restrict__ wsB,      // packed B fragments
    float*       __restrict__ out) {    // ss [T][B][COUT][H][W] ++ mem_out

  __shared__ bf16  sB[KSTEPS * 4 * 32 * 16];     // 72 KB
  __shared__ bf16  sA2[2][A_ELEMS];              // 2 x 17 KB, double buffered
  __shared__ float sOut[TILE_N * SOUT_STRIDE];   // 17 KB
  __shared__ bf16  sDummy[8];                    // sink for OOB async lanes

  const int tid   = threadIdx.x;
  const int bid   = blockIdx.x;
  const int b     = bid >> 5;
  const int tileY = (bid >> 1) & 15;
  const int nHalf = bid & 1;
  const int coutBase = nHalf * TILE_N;

  const int wave = tid >> 5;
  const int lane = tid & 31;
  const int half = lane >> 4;
  const int ln   = lane & 15;
  const int mT   = wave & 3;
  const int nTl0 = (wave >> 2) * 2;

  // ---- stage this WG's B fragments to LDS (reused KSTEPS*T times) ----------
  for (int c = tid; c < KSTEPS * 256; c += 256) {
    int ks = c >> 8;
    int i  = (c & 255) * 8;
    *(uint4*)&sB[ks * 2048 + i] =
        *(const uint4*)&wsB[ks * 4096 + nHalf * 2048 + i];
  }

  // ---- per-lane membrane state + bias --------------------------------------
  v8f   memv[2];
  float biasv[2];
  const int mbase = mT * 16 + half * 8;
  for (int q = 0; q < 2; ++q) {
    int cout = coutBase + (nTl0 + q) * 16 + ln;
    biasv[q] = bias[cout];
    for (int r = 0; r < 8; ++r) {
      int pix = mbase + r;
      int y = tileY * TILE_H + (pix >> 5);
      int x = pix & 31;
      memv[q][r] = mem_in[((b * COUT + cout) * H_ + y) * W_ + x];
    }
  }

  // ---- zero constant halo borders in BOTH buffers (never touched by DMA) ---
  for (int v = tid; v < 136 * 8; v += 256) {
    int pix = v >> 3, c = v & 7;
    int ry = pix / 34, rx = pix - ry * 34;
    int y = tileY * TILE_H + ry - 1;
    if (rx == 0 || rx == 33 || (unsigned)y >= (unsigned)H_) {
      uint4 z = {0u, 0u, 0u, 0u};
      *(uint4*)&sA2[0][pix * CIN + c * 8] = z;
      *(uint4*)&sA2[1][pix * CIN + c * 8] = z;
    }
  }

  const unsigned aBase0   = (unsigned)(uintptr_t)&sA2[0][0];
  const unsigned aBase1   = (unsigned)(uintptr_t)&sA2[1][0];
  const unsigned dummyOff = (unsigned)(uintptr_t)&sDummy[0];

  // Issue one tile's async copy: 128 interior pixels x 8 b128 chunks = 1024
  // per-lane transfers -> exactly 4 async instructions per wave (OOB lanes are
  // redirected to sDummy so EXEC stays full and ASYNCcnt bookkeeping is exact).
  auto issue_tile = [&](int t, int bufSel) {
    const bf16* base = wsS + (size_t)(t * B_ + b) * (H_ * W_ * CIN);
    const unsigned aB = bufSel ? aBase1 : aBase0;
    for (int it = 0; it < 4; ++it) {
      int v   = it * 256 + tid;
      int pix = v >> 3;                 // 0..127, rx = (pix&31)+1
      int c   = v & 7;
      int ry  = pix >> 5;
      int x   = pix & 31;
      int y   = tileY * TILE_H + ry - 1;
      bool valid = (unsigned)y < (unsigned)H_;
      int yc = valid ? y : 0;
      unsigned lds = valid
          ? aB + (unsigned)(((ry * HALO_W + x + 1) * CIN + c * 8) * 2)
          : dummyOff;
      unsigned long long ga =
          (unsigned long long)(uintptr_t)(base + ((size_t)yc * W_ + x) * CIN + c * 8);
      asm volatile("global_load_async_to_lds_b128 %0, %1, off"
                   :: "v"(lds), "v"(ga) : "memory");
    }
  };

  issue_tile(0, 0);

  const int    pixA = mT * 16 + ln;
  const int    oyA  = pixA >> 5, oxA = pixA & 31;

  for (int t = 0; t < T_; ++t) {
    const int cur = t & 1;
    if (t + 1 < T_) {
      issue_tile(t + 1, cur ^ 1);       // prefetch next tile into other buffer
#if __has_builtin(__builtin_amdgcn_s_wait_asynccnt)
      __builtin_amdgcn_s_wait_asynccnt(4);   // keep next tile's 4 in flight
#else
      asm volatile("s_wait_asynccnt 0x4" ::: "memory");
#endif
    } else {
#if __has_builtin(__builtin_amdgcn_s_wait_asynccnt)
      __builtin_amdgcn_s_wait_asynccnt(0);
#else
      asm volatile("s_wait_asynccnt 0x0" ::: "memory");
#endif
    }
    __syncthreads();                    // tile t visible to all waves

    const bf16* aBuf = &sA2[cur][0];

    v8f acc[2];
    for (int q = 0; q < 2; ++q)
      for (int r = 0; r < 8; ++r)
        acc[q][r] = memv[q][r] + biasv[q];

    for (int kh = 0; kh < 3; ++kh)
      for (int kw = 0; kw < 3; ++kw)
        for (int kHalf = 0; kHalf < 2; ++kHalf) {
          int aoff = ((oyA + kh) * HALO_W + (oxA + kw)) * CIN
                     + kHalf * 32 + half * 16;
          v16bf aF = *(const v16bf*)&aBuf[aoff];
          int ks = (kh * 3 + kw) * 2 + kHalf;
          const bf16* bp = &sB[ks * 2048 + lane * 16];
          v16bf b0 = *(const v16bf*)&bp[(nTl0 + 0) * 512];
          v16bf b1 = *(const v16bf*)&bp[(nTl0 + 1) * 512];
          acc[0] = __builtin_amdgcn_wmma_f32_16x16x32_bf16(
              false, aF, false, b0, (short)0, acc[0], false, false);
          acc[1] = __builtin_amdgcn_wmma_f32_16x16x32_bf16(
              false, aF, false, b1, (short)0, acc[1], false, false);
        }

    for (int q = 0; q < 2; ++q) {
      int cl = (nTl0 + q) * 16 + ln;
      for (int r = 0; r < 8; ++r) {
        float m = acc[q][r];
        float s = (m > VTH) ? 1.0f : 0.0f;
        m = fmaxf(m, 0.0f);
        m = m - m * s;
        memv[q][r] = m;
        sOut[cl * SOUT_STRIDE + mbase + r] = s;
      }
    }
    __syncthreads();                    // all tile-t reads + sOut writes done

    size_t obase = ((size_t)(t * B_ + b) * COUT + coutBase) * (H_ * W_)
                   + (size_t)tileY * TILE_H * W_;
    for (int c = tid; c < 1024; c += 256) {
      int cl = c >> 4;
      int f  = (c & 15) << 2;
      *(float4*)&out[obase + (size_t)cl * (H_ * W_) + f] =
          *(const float4*)&sOut[cl * SOUT_STRIDE + f];
    }
  }

  __syncthreads();
  for (int q = 0; q < 2; ++q) {
    int cl = (nTl0 + q) * 16 + ln;
    for (int r = 0; r < 8; ++r)
      sOut[cl * SOUT_STRIDE + mbase + r] = memv[q][r];
  }
  __syncthreads();
  size_t mo = (size_t)T_ * B_ * COUT * H_ * W_
            + ((size_t)b * COUT + coutBase) * (H_ * W_)
            + (size_t)tileY * TILE_H * W_;
  for (int c = tid; c < 1024; c += 256) {
    int cl = c >> 4;
    int f  = (c & 15) << 2;
    *(float4*)&out[mo + (size_t)cl * (H_ * W_) + f] =
        *(const float4*)&sOut[cl * SOUT_STRIDE + f];
  }
}

// ---------------------------------------------------------------------------
// Fallback fused kernel (direct f32 loads, no workspace for spikes).
// ---------------------------------------------------------------------------
__global__ __launch_bounds__(256) void snn_conv_scan(
    const float* __restrict__ spikes, const float* __restrict__ mem_in,
    const float* __restrict__ bias, const bf16* __restrict__ wsB,
    float* __restrict__ out) {

  __shared__ bf16  sB[KSTEPS * 4 * 32 * 16];
  __shared__ bf16  sA[A_ELEMS];
  __shared__ float sOut[TILE_N * SOUT_STRIDE];

  const int tid   = threadIdx.x;
  const int bid   = blockIdx.x;
  const int b     = bid >> 5;
  const int tileY = (bid >> 1) & 15;
  const int nHalf = bid & 1;
  const int coutBase = nHalf * TILE_N;

  const int wave = tid >> 5;
  const int lane = tid & 31;
  const int half = lane >> 4;
  const int ln   = lane & 15;
  const int mT   = wave & 3;
  const int nTl0 = (wave >> 2) * 2;

  for (int c = tid; c < KSTEPS * 256; c += 256) {
    int ks = c >> 8;
    int i  = (c & 255) * 8;
    *(uint4*)&sB[ks * 2048 + i] =
        *(const uint4*)&wsB[ks * 4096 + nHalf * 2048 + i];
  }

  v8f   memv[2];
  float biasv[2];
  const int mbase = mT * 16 + half * 8;
  for (int q = 0; q < 2; ++q) {
    int cout = coutBase + (nTl0 + q) * 16 + ln;
    biasv[q] = bias[cout];
    for (int r = 0; r < 8; ++r) {
      int pix = mbase + r;
      int y = tileY * TILE_H + (pix >> 5);
      int x = pix & 31;
      memv[q][r] = mem_in[((b * COUT + cout) * H_ + y) * W_ + x];
    }
  }
  __syncthreads();

  const int    pixA = mT * 16 + ln;
  const int    oyA  = pixA >> 5, oxA = pixA & 31;
  const size_t tStride = (size_t)B_ * CIN * H_ * W_;

  for (int t = 0; t < T_; ++t) {
    const float* sp = spikes + (size_t)(t * B_ + b) * CIN * H_ * W_;
    for (int it = 0; it < 34; ++it) {
      int e   = it * 256 + tid;
      int cin = e / 136;
      int p   = e - cin * 136;
      int ry  = p / HALO_W;
      int rx  = p - ry * HALO_W;
      int y   = tileY * TILE_H + ry - 1;
      int x   = rx - 1;
      float v = 0.0f;
      if ((unsigned)y < (unsigned)H_ && (unsigned)x < (unsigned)W_) {
        const float* g = sp + ((size_t)cin * H_ + y) * W_ + x;
        v = *g;
        if (t + 1 < T_) __builtin_prefetch(g + tStride, 0, 1);
      }
      sA[(ry * HALO_W + rx) * CIN + cin] = (bf16)v;
    }
    __syncthreads();

    v8f acc[2];
    for (int q = 0; q < 2; ++q)
      for (int r = 0; r < 8; ++r)
        acc[q][r] = memv[q][r] + biasv[q];

    for (int kh = 0; kh < 3; ++kh)
      for (int kw = 0; kw < 3; ++kw)
        for (int kHalf = 0; kHalf < 2; ++kHalf) {
          int aoff = ((oyA + kh) * HALO_W + (oxA + kw)) * CIN
                     + kHalf * 32 + half * 16;
          v16bf aF = *(const v16bf*)&sA[aoff];
          int ks = (kh * 3 + kw) * 2 + kHalf;
          const bf16* bp = &sB[ks * 2048 + lane * 16];
          v16bf b0 = *(const v16bf*)&bp[(nTl0 + 0) * 512];
          v16bf b1 = *(const v16bf*)&bp[(nTl0 + 1) * 512];
          acc[0] = __builtin_amdgcn_wmma_f32_16x16x32_bf16(
              false, aF, false, b0, (short)0, acc[0], false, false);
          acc[1] = __builtin_amdgcn_wmma_f32_16x16x32_bf16(
              false, aF, false, b1, (short)0, acc[1], false, false);
        }

    for (int q = 0; q < 2; ++q) {
      int cl = (nTl0 + q) * 16 + ln;
      for (int r = 0; r < 8; ++r) {
        float m = acc[q][r];
        float s = (m > VTH) ? 1.0f : 0.0f;
        m = fmaxf(m, 0.0f);
        m = m - m * s;
        memv[q][r] = m;
        sOut[cl * SOUT_STRIDE + mbase + r] = s;
      }
    }
    __syncthreads();

    size_t obase = ((size_t)(t * B_ + b) * COUT + coutBase) * (H_ * W_)
                   + (size_t)tileY * TILE_H * W_;
    for (int c = tid; c < 1024; c += 256) {
      int cl = c >> 4;
      int f  = (c & 15) << 2;
      *(float4*)&out[obase + (size_t)cl * (H_ * W_) + f] =
          *(const float4*)&sOut[cl * SOUT_STRIDE + f];
    }
  }

  __syncthreads();
  for (int q = 0; q < 2; ++q) {
    int cl = (nTl0 + q) * 16 + ln;
    for (int r = 0; r < 8; ++r)
      sOut[cl * SOUT_STRIDE + mbase + r] = memv[q][r];
  }
  __syncthreads();
  size_t mo = (size_t)T_ * B_ * COUT * H_ * W_
            + ((size_t)b * COUT + coutBase) * (H_ * W_)
            + (size_t)tileY * TILE_H * W_;
  for (int c = tid; c < 1024; c += 256) {
    int cl = c >> 4;
    int f  = (c & 15) << 2;
    *(float4*)&out[mo + (size_t)cl * (H_ * W_) + f] =
        *(const float4*)&sOut[cl * SOUT_STRIDE + f];
  }
}

// ---------------------------------------------------------------------------
extern "C" void kernel_launch(void* const* d_in, const int* in_sizes, int n_in,
                              void* d_out, int out_size, void* d_ws,
                              size_t ws_size, hipStream_t stream) {
  const float* spikes = (const float*)d_in[0];
  const float* mem    = (const float*)d_in[1];
  const float* weight = (const float*)d_in[2];
  const float* bias   = (const float*)d_in[3];
  float* out = (float*)d_out;

  const size_t needAsync = WSS_ELEMS * 2 + (size_t)WSB_ELEMS * 2;

  if (ws_size >= needAsync) {
    bf16* wsS = (bf16*)d_ws;
    bf16* wsB = (bf16*)((char*)d_ws + WSS_ELEMS * 2);
    snn_pack_weights<<<(WSB_ELEMS + 255) / 256, 256, 0, stream>>>(weight, wsB);
    snn_transpose_spikes<<<T_ * B_ * H_, 256, 0, stream>>>(spikes, wsS);
    snn_conv_scan_async<<<B_ * 16 * 2, 256, 0, stream>>>(wsS, mem, bias, wsB, out);
  } else {
    bf16* wsB = (bf16*)d_ws;
    snn_pack_weights<<<(WSB_ELEMS + 255) / 256, 256, 0, stream>>>(weight, wsB);
    snn_conv_scan<<<B_ * 16 * 2, 256, 0, stream>>>(spikes, mem, bias, wsB, out);
  }
}
